// MultiHeadAttention_22960895165035
// MI455X (gfx1250) — matrix operand order repaired and checked by reading, not compiled, not run on hardware
//
#include <hip/hip_runtime.h>
#include <hip/hip_bf16.h>

// ---------------------------------------------------------------------------
// Multi-head attention for MI455X (gfx1250, wave32, WMMA).
//   D_MODEL=1024, d_k=64, H=16, N=2, T=2048  -> ROWS = 4096
// Pipeline: 3x projection GEMM (fp32 in -> bf16 out, head-split),
//           flash-attention (bf16 WMMA, fp32 softmax, double-buffered
//           GLOBAL_LOAD_ASYNC_TO_LDS_B128), 1x output GEMM (bf16 -> fp32).
// Workspace: 4 x 4096x1024 bf16 buffers = 32 MB.
// ---------------------------------------------------------------------------

#define D_MODEL 1024
#define D_K     64
#define N_HEADS 16
#define NB      2
#define TSEQ    2048
#define ROWS    (NB * TSEQ)   // 4096

typedef __attribute__((ext_vector_type(16))) __bf16 v16bf;
typedef __attribute__((ext_vector_type(8)))  float  v8f;

union FragBF {
    v16bf  v;
    uint4  u[2];
    __bf16 h[16];
};
union PairBF { __bf16 h[2]; unsigned u; };

// fp32 -> bf16 via native convert (v_cvt_*bf16_f32 on gfx1250, RNE)
__device__ inline __bf16 f2bf(float x) { return (__bf16)x; }

__device__ inline v8f zero8() {
    v8f z;
    #pragma unroll
    for (int i = 0; i < 8; ++i) z[i] = 0.f;
    return z;
}

// Async global->LDS 16B copy (CDNA5, tracked by ASYNCcnt, in-order done).
__device__ inline void async_ld_b128(unsigned ldsOff, const void* gaddr) {
    asm volatile("global_load_async_to_lds_b128 %0, %1, off"
                 :: "v"(ldsOff), "v"(gaddr) : "memory");
}
__device__ inline void wait_async0() {  // all async ops done
    asm volatile("s_wait_asynccnt 0x0" ::: "memory");
}
__device__ inline void wait_async4() {  // <=4 outstanding (prev tile done)
    asm volatile("s_wait_asynccnt 0x4" ::: "memory");
}
__device__ inline unsigned lds_off(const void* p) {
    return (unsigned)(size_t)p;   // low 32 bits = LDS byte offset
}

// ---------------------------------------------------------------------------
// GEMM: Y = X(4096x1024) * W(1024x1024) + bias
// Block tile 128x128, K-step 32, 8 waves (each 32x64 = 2x4 WMMA tiles).
// Software-pipelined: next k-tile is fetched into registers during WMMA.
// EPI 0: fp32 row-major out.  EPI 1: bf16 head-split [b,h,t,d].
// EPI 2: bf16 head-split transposed [b,h,d,t]  (for V).
// ---------------------------------------------------------------------------
template<int EPI, bool XBF16>
__global__ __launch_bounds__(256)
void mha_gemm(const void* __restrict__ Xv, const float* __restrict__ W,
              const float* __restrict__ bias, void* __restrict__ Yv)
{
    // padded row strides (40 elems = 80 B) -> conflict-free 16-lane frag reads
    __shared__ __bf16 As[128][40];   // [row][k]
    __shared__ __bf16 Bs[128][40];   // [col][k] (W transposed on store)

    const int tid  = threadIdx.x;
    const int lane = tid & 31;
    const int wid  = tid >> 5;
    const int half = lane >> 4;
    const int l16  = lane & 15;

    const int rowBase = blockIdx.y * 128;
    const int colBase = blockIdx.x * 128;
    const int mBase   = (wid >> 1) * 32;  // 0,32,64,96
    const int nBase   = (wid & 1) * 64;   // 0,64

    v8f acc[2][4];
    #pragma unroll
    for (int i = 0; i < 2; ++i)
        #pragma unroll
        for (int j = 0; j < 4; ++j) acc[i][j] = zero8();

    const int ldRow = tid >> 1;         // 0..127  (A-tile row)
    const int ldSeg = (tid & 1) * 16;   // 0 or 16 (A-tile k segment)
    const int bN    = tid >> 1;         // 0..127  (B-tile col)
    const int bKs   = (tid & 1) * 16;   // 0 or 16 (B-tile k segment)

    // register staging for the next k-tile
    float aF[16];
    uint4 aU[2];
    float bF[16];

    auto load_regs = [&](int k0) {
        if constexpr (XBF16) {
            const __bf16* X = (const __bf16*)Xv;
            const uint4* src =
                (const uint4*)(X + (size_t)(rowBase + ldRow) * D_MODEL + k0 + ldSeg);
            aU[0] = src[0]; aU[1] = src[1];
        } else {
            const float* X = (const float*)Xv;
            const float4* src =
                (const float4*)(X + (size_t)(rowBase + ldRow) * D_MODEL + k0 + ldSeg);
            float4 f0 = src[0], f1 = src[1], f2 = src[2], f3 = src[3];
            aF[0]=f0.x; aF[1]=f0.y; aF[2]=f0.z;  aF[3]=f0.w;
            aF[4]=f1.x; aF[5]=f1.y; aF[6]=f1.z;  aF[7]=f1.w;
            aF[8]=f2.x; aF[9]=f2.y; aF[10]=f2.z; aF[11]=f2.w;
            aF[12]=f3.x;aF[13]=f3.y;aF[14]=f3.z; aF[15]=f3.w;
        }
        #pragma unroll
        for (int i = 0; i < 16; ++i)
            bF[i] = W[(size_t)(k0 + bKs + i) * D_MODEL + colBase + bN];
        if (k0 + 32 < D_MODEL)   // speculative prefetch one more tile ahead
            __builtin_prefetch(&W[(size_t)(k0 + 32 + bKs) * D_MODEL + colBase + bN], 0, 0);
    };

    load_regs(0);

    for (int k0 = 0; k0 < D_MODEL; k0 += 32) {
        // ---- stage registers -> LDS (bf16) ----
        {
            FragBF pk;
            if constexpr (XBF16) {
                pk.u[0] = aU[0]; pk.u[1] = aU[1];
            } else {
                #pragma unroll
                for (int i = 0; i < 16; ++i) pk.h[i] = f2bf(aF[i]);
            }
            uint4* dst = (uint4*)&As[ldRow][ldSeg];
            dst[0] = pk.u[0]; dst[1] = pk.u[1];

            #pragma unroll
            for (int j = 0; j < 8; ++j) {            // packed b32 stores
                PairBF pr;
                pr.h[0] = f2bf(bF[2 * j]);
                pr.h[1] = f2bf(bF[2 * j + 1]);
                *(unsigned*)&Bs[bN][bKs + 2 * j] = pr.u;
            }
        }
        __syncthreads();

        // ---- fetch next tile into registers (overlaps with WMMA below) ----
        if (k0 + 32 < D_MODEL) load_regs(k0 + 32);

        // ---- fragments + WMMA ----
        FragBF a[2];
        #pragma unroll
        for (int mi = 0; mi < 2; ++mi) {
            // A 16x32 bf16 layout: lanes 0-15 rows, K 0-7 & 16-23; lanes 16-31 K 8-15 & 24-31
            a[mi].u[0] = *(const uint4*)&As[mBase + mi * 16 + l16][half * 8];
            a[mi].u[1] = *(const uint4*)&As[mBase + mi * 16 + l16][16 + half * 8];
        }
        #pragma unroll
        for (int nj = 0; nj < 4; ++nj) {
            FragBF b;
            // B 32x16 layout: lanes 0-15 K 0-7 & 16-23, lanes 16-31 K 8-15 & 24-31, N = lane%16
            b.u[0] = *(const uint4*)&Bs[nBase + nj * 16 + l16][half * 8];
            b.u[1] = *(const uint4*)&Bs[nBase + nj * 16 + l16][16 + half * 8];
            #pragma unroll
            for (int mi = 0; mi < 2; ++mi) {
                acc[mi][nj] = __builtin_amdgcn_wmma_f32_16x16x32_bf16(
                    false, a[mi].v, false, b.v, (short)0, acc[mi][nj], false, false);
            }
        }
        __syncthreads();
    }

    // ---- epilogue ----
    #pragma unroll
    for (int mi = 0; mi < 2; ++mi) {
        #pragma unroll
        for (int nj = 0; nj < 4; ++nj) {
            const int col = colBase + nBase + nj * 16 + l16;
            const float bv = bias[col];
            #pragma unroll
            for (int r = 0; r < 8; ++r) {
                const int row = rowBase + mBase + mi * 16 + half * 8 + r;
                const float val = acc[mi][nj][r] + bv;
                if constexpr (EPI == 0) {
                    ((float*)Yv)[(size_t)row * D_MODEL + col] = val;
                } else if constexpr (EPI == 1) {
                    const int bh = (row / TSEQ) * N_HEADS + (col >> 6);
                    const int t  = row % TSEQ;
                    const int d  = col & 63;
                    ((__bf16*)Yv)[((size_t)bh * TSEQ + t) * D_K + d] = f2bf(val);
                } else {
                    const int bh = (row / TSEQ) * N_HEADS + (col >> 6);
                    const int t  = row % TSEQ;
                    const int d  = col & 63;
                    ((__bf16*)Yv)[((size_t)bh * D_K + d) * TSEQ + t] = f2bf(val);
                }
            }
        }
    }
}

// ---------------------------------------------------------------------------
// Flash attention: one block = 128 queries of one (b,h); 8 waves x 16 rows.
// 64-key tiles double-buffered via GLOBAL_LOAD_ASYNC_TO_LDS_B128 (ASYNCcnt
// in-order completion: wait <=4 after issuing next tile's 4 ops).
// Q,K in [b,h,t,d] bf16; V in [b,h,d,t] bf16; out A in [b,t,h*64+d] bf16.
// ---------------------------------------------------------------------------
__global__ __launch_bounds__(256)
void mha_attn(const __bf16* __restrict__ Qb, const __bf16* __restrict__ Kb,
              const __bf16* __restrict__ Vt, const int* __restrict__ mask,
              __bf16* __restrict__ Ab)
{
    __shared__ __bf16 Ks[2][64][72];     // [buf][key][d]  (144B stride)
    __shared__ __bf16 Vs[2][64][72];     // [buf][d][key]
    __shared__ __bf16 Ps[8][16][72];     // per-wave P staging: [row][key]

    const int tid  = threadIdx.x;
    const int lane = tid & 31;
    const int wid  = tid >> 5;
    const int half = lane >> 4;
    const int l16  = lane & 15;

    const int bh    = blockIdx.y;
    const int b     = bh >> 4;
    const int h     = bh & 15;
    const int qBase = blockIdx.x * 128 + wid * 16;

    // Q fragments held in registers (16 rows x 64 d = two K=32 A-frags)
    FragBF qf[2];
    {
        const __bf16* qrow = Qb + ((size_t)bh * TSEQ + qBase + l16) * D_K;
        #pragma unroll
        for (int f = 0; f < 2; ++f) {
            qf[f].u[0] = *(const uint4*)(qrow + f * 32 + half * 8);
            qf[f].u[1] = *(const uint4*)(qrow + f * 32 + 16 + half * 8);
        }
    }

    float mrow[8], lrow[8];
    v8f o[4];
    #pragma unroll
    for (int r = 0; r < 8; ++r) { mrow[r] = -3.0e38f; lrow[r] = 0.f; }
    #pragma unroll
    for (int dt = 0; dt < 4; ++dt) o[dt] = zero8();

    const int ldr = tid >> 2;          // 0..63
    const int lds = (tid & 3) * 16;    // 0,16,32,48

    const unsigned ksOff[2] = { lds_off(&Ks[0][ldr][lds]), lds_off(&Ks[1][ldr][lds]) };
    const unsigned vsOff[2] = { lds_off(&Vs[0][ldr][lds]), lds_off(&Vs[1][ldr][lds]) };

    auto issue_tile = [&](int kBase, int buf) {
        const __bf16* kg = Kb + ((size_t)bh * TSEQ + kBase + ldr) * D_K + lds;
        async_ld_b128(ksOff[buf],      kg);
        async_ld_b128(ksOff[buf] + 16, kg + 8);
        const __bf16* vg = Vt + ((size_t)bh * D_K + ldr) * TSEQ + kBase + lds;
        async_ld_b128(vsOff[buf],      vg);
        async_ld_b128(vsOff[buf] + 16, vg + 8);
    };

    issue_tile(0, 0);

    for (int it = 0; it < TSEQ / 64; ++it) {
        const int kBase = it * 64;
        const int cur   = it & 1;

        if (kBase + 64 < TSEQ) {           // prefetch next tile, then wait for
            issue_tile(kBase + 64, cur ^ 1);  // current one (in-order: cnt<=4)
            wait_async4();
        } else {
            wait_async0();
        }
        __syncthreads();

        // ---- scores S = Q K^T (raw; scale folded into exp) ----
        v8f s[4];
        #pragma unroll
        for (int kt = 0; kt < 4; ++kt) {
            v8f sacc = zero8();
            #pragma unroll
            for (int df = 0; df < 2; ++df) {
                FragBF bk;
                bk.u[0] = *(const uint4*)&Ks[cur][kt * 16 + l16][df * 32 + half * 8];
                bk.u[1] = *(const uint4*)&Ks[cur][kt * 16 + l16][df * 32 + 16 + half * 8];
                sacc = __builtin_amdgcn_wmma_f32_16x16x32_bf16(
                    false, qf[df].v, false, bk.v, (short)0, sacc, false, false);
            }
            s[kt] = sacc;
        }

        // ---- mask (lane holds one key column per tile) ----
        #pragma unroll
        for (int kt = 0; kt < 4; ++kt) {
            const float madd =
                mask[(size_t)b * TSEQ + kBase + kt * 16 + l16] ? 0.f : -3.0e38f;
            #pragma unroll
            for (int r = 0; r < 8; ++r) s[kt][r] = s[kt][r] + madd;
        }

        // ---- row max across 4 tiles (in-lane) + 16 lanes of this half ----
        float rm[8];
        #pragma unroll
        for (int r = 0; r < 8; ++r) {
            float m = s[0][r];
            m = fmaxf(m, s[1][r]); m = fmaxf(m, s[2][r]); m = fmaxf(m, s[3][r]);
            rm[r] = m;
        }
        #pragma unroll
        for (int off = 8; off >= 1; off >>= 1)
            #pragma unroll
            for (int r = 0; r < 8; ++r)
                rm[r] = fmaxf(rm[r], __shfl_xor(rm[r], off, 32));

        float scaleO[8];
        #pragma unroll
        for (int r = 0; r < 8; ++r) {
            const float mn = fmaxf(mrow[r], rm[r]);
            scaleO[r] = __expf((mrow[r] - mn) * 0.125f);  // 1/sqrt(64)
            mrow[r] = mn;
        }
        #pragma unroll
        for (int dt = 0; dt < 4; ++dt)
            #pragma unroll
            for (int r = 0; r < 8; ++r) o[dt][r] = o[dt][r] * scaleO[r];

        // ---- P = exp((S-m)/8): write bf16 to per-wave LDS in [row][key] ----
        float rs[8];
        #pragma unroll
        for (int r = 0; r < 8; ++r) rs[r] = 0.f;
        #pragma unroll
        for (int kt = 0; kt < 4; ++kt) {
            #pragma unroll
            for (int r = 0; r < 8; ++r) {
                const float p = __expf((s[kt][r] - mrow[r]) * 0.125f);
                rs[r] += p;
                Ps[wid][half * 8 + r][kt * 16 + l16] = f2bf(p);
            }
        }
        #pragma unroll
        for (int off = 8; off >= 1; off >>= 1)
            #pragma unroll
            for (int r = 0; r < 8; ++r)
                rs[r] += __shfl_xor(rs[r], off, 32);
        #pragma unroll
        for (int r = 0; r < 8; ++r) lrow[r] = lrow[r] * scaleO[r] + rs[r];

        // ---- O += P V  (P re-read from own-wave LDS in A-layout) ----
        #pragma unroll
        for (int kf = 0; kf < 2; ++kf) {
            FragBF pa;
            pa.u[0] = *(const uint4*)&Ps[wid][l16][kf * 32 + half * 8];
            pa.u[1] = *(const uint4*)&Ps[wid][l16][kf * 32 + 16 + half * 8];
            #pragma unroll
            for (int dt = 0; dt < 4; ++dt) {
                FragBF vb;
                vb.u[0] = *(const uint4*)&Vs[cur][dt * 16 + l16][kf * 32 + half * 8];
                vb.u[1] = *(const uint4*)&Vs[cur][dt * 16 + l16][kf * 32 + 16 + half * 8];
                o[dt] = __builtin_amdgcn_wmma_f32_16x16x32_bf16(
                    false, pa.v, false, vb.v, (short)0, o[dt], false, false);
            }
        }
        __syncthreads();
    }

    // ---- epilogue: O / l -> bf16 head-merged [b, t, h*64+d] ----
    #pragma unroll
    for (int dt = 0; dt < 4; ++dt) {
        #pragma unroll
        for (int r = 0; r < 8; ++r) {
            const int t   = qBase + half * 8 + r;
            const int col = h * 64 + dt * 16 + l16;
            const float val = o[dt][r] * (1.0f / lrow[r]);
            Ab[((size_t)b * TSEQ + t) * D_MODEL + col] = f2bf(val);
        }
    }
}

// ---------------------------------------------------------------------------
extern "C" void kernel_launch(void* const* d_in, const int* in_sizes, int n_in,
                              void* d_out, int out_size, void* d_ws, size_t ws_size,
                              hipStream_t stream)
{
    (void)in_sizes; (void)n_in; (void)out_size; (void)ws_size;

    const float* k    = (const float*)d_in[0];
    const float* q    = (const float*)d_in[1];
    const float* v    = (const float*)d_in[2];
    const int*   mask = (const int*)  d_in[3];
    const float* Wk   = (const float*)d_in[4];
    const float* bk   = (const float*)d_in[5];
    const float* Wq   = (const float*)d_in[6];
    const float* bq   = (const float*)d_in[7];
    const float* Wv   = (const float*)d_in[8];
    const float* bv   = (const float*)d_in[9];
    const float* Wo   = (const float*)d_in[10];
    const float* bo   = (const float*)d_in[11];
    float* out = (float*)d_out;

    // workspace: 4 x (4096*1024) bf16 = 32 MB
    __bf16* Qb = (__bf16*)d_ws;
    __bf16* Kb = Qb + (size_t)ROWS * D_MODEL;
    __bf16* Vt = Kb + (size_t)ROWS * D_MODEL;
    __bf16* Ab = Vt + (size_t)ROWS * D_MODEL;

    dim3 gg(D_MODEL / 128, ROWS / 128);   // 8 x 32 = 256 workgroups
    mha_gemm<1, false><<<gg, 256, 0, stream>>>(q, Wq, bq, Qb);
    mha_gemm<1, false><<<gg, 256, 0, stream>>>(k, Wk, bk, Kb);
    mha_gemm<2, false><<<gg, 256, 0, stream>>>(v, Wv, bv, Vt);

    dim3 ga(TSEQ / 128, NB * N_HEADS);    // 16 x 32 = 512 workgroups
    mha_attn<<<ga, 256, 0, stream>>>(Qb, Kb, Vt, mask, Ab);

    mha_gemm<0, true><<<gg, 256, 0, stream>>>(Ab, Wo, bo, out);
}
